// MambaBlock_33775622815898
// MI455X (gfx1250) — matrix operand order repaired and checked by reading, not compiled, not run on hardware
//
#include <hip/hip_runtime.h>
#include <hip/hip_bf16.h>

// ---------------------------------------------------------------------------
// Mamba block for MI455X (gfx1250, wave32, WMMA).
// Shapes: B=4, L=2048, D_MODEL=1024, D_INNER=2048, D_STATE=64, D_CONV=4, DT_RANK=64
// ---------------------------------------------------------------------------

#define B_SZ     4
#define L_SZ     2048
#define D_MODEL  1024
#define D_INNER  2048
#define D_STATE  64
#define D_CONV   4
#define DT_RANK  64
#define ROWS     (B_SZ * L_SZ)            // 8192 tokens
#define XPROJ_N  (DT_RANK + 2 * D_STATE)  // 192

typedef __attribute__((ext_vector_type(16))) _Float16 v16h;
typedef __attribute__((ext_vector_type(8)))  float    v8f;

union HFrag {
    v16h   v;
    _Float16 h[16];
    float4 q[2];   // two 16-byte chunks
};

__device__ __forceinline__ float fast_sigmoid(float x) {
    return 1.0f / (1.0f + __expf(-x));
}

// ---------------------------------------------------------------------------
// f32 -> f16 convert (weights / activations)
// ---------------------------------------------------------------------------
__global__ __launch_bounds__(256) void cvt_f16_kernel(const float* __restrict__ in,
                                                      _Float16* __restrict__ out, int n) {
    int i = blockIdx.x * 256 + threadIdx.x;
    if (i < n) out[i] = (_Float16)in[i];
}

// ---------------------------------------------------------------------------
// RMSNorm: one block per token row of 1024, writes f16 for WMMA GEMM
// ---------------------------------------------------------------------------
__global__ __launch_bounds__(256) void rmsnorm_kernel(const float* __restrict__ x,
                                                      const float* __restrict__ w,
                                                      _Float16* __restrict__ out) {
    __shared__ float red[256];
    const int row = blockIdx.x;
    const float* xr = x + (size_t)row * D_MODEL;
    float s = 0.f;
    for (int i = threadIdx.x; i < D_MODEL; i += 256) {
        float v = xr[i];
        s = fmaf(v, v, s);
    }
    red[threadIdx.x] = s;
    __syncthreads();
    for (int off = 128; off > 0; off >>= 1) {
        if (threadIdx.x < off) red[threadIdx.x] += red[threadIdx.x + off];
        __syncthreads();
    }
    const float scale = rsqrtf(red[0] * (1.0f / D_MODEL) + 1.1920928955078125e-07f);
    for (int i = threadIdx.x; i < D_MODEL; i += 256) {
        out[(size_t)row * D_MODEL + i] = (_Float16)(xr[i] * scale * w[i]);
    }
}

// ---------------------------------------------------------------------------
// WMMA GEMM: C[M,N] = A[M,K] (f16, row-major) x Bw[N,K]^T (f16, row-major)
// One wave -> 16 rows x 64 cols (A fragment reused across 4 WMMAs).
// Block = 128 threads = 4 waves stacked along M -> 64x64 block tile.
// Software-pipelined K loop (double-buffered fragments) so WMMAs overlap
// the next iteration's global loads instead of draining loadcnt to 0.
// MODE 0: store; MODE 1: softplus(acc + bias[col]); MODE 2: acc + R[row,col]
// ---------------------------------------------------------------------------
template <int MODE>
__global__ __launch_bounds__(128)
void gemm_wmma_f16(const _Float16* __restrict__ A, const _Float16* __restrict__ Bw,
                   float* __restrict__ C, const float* __restrict__ extra,
                   int M, int N, int K) {
    const int lane    = threadIdx.x & 31;
    const int wave    = threadIdx.x >> 5;
    const int lane15  = lane & 15;
    const int halfSel = lane >> 4;                       // 0 | 1
    const int tileM   = (blockIdx.y * 4 + wave) * 16;
    const int tileN   = blockIdx.x * 64;

    // A-fragment addressing (16-bit A 16x32 layout, ISA 7.12.2):
    //   lane<16 : row=lane,     K chunks {k0..k0+7, k0+16..k0+23}
    //   lane>=16: row=lane-16,  K chunks {k0+8..k0+15, k0+24..k0+31}
    const _Float16* aRow = A + (size_t)(tileM + lane15) * K + halfSel * 8;
    // B-fragment addressing (32x16): lane<16 -> col=lane, K {0..7,16..23}; lane>=16 -> K {8..15,24..31}
    const int colBase = tileN + lane15;
    const int kOffB   = halfSel * 8;

    v8f acc[4] = {};

    HFrag af, bf[4];
    // ---- prologue: load fragments for k0 = 0 ----
    af.q[0] = *(const float4*)(aRow);
    af.q[1] = *(const float4*)(aRow + 16);
#pragma unroll
    for (int j = 0; j < 4; ++j) {
        const _Float16* bp = Bw + (size_t)(colBase + 16 * j) * K + kOffB;
        bf[j].q[0] = *(const float4*)(bp);
        bf[j].q[1] = *(const float4*)(bp + 16);
    }

    int k0 = 0;
    for (; k0 + 32 < K; k0 += 32) {
        // ---- issue next-iteration loads first (stay in flight across WMMAs)
        const int kn = k0 + 32;
        HFrag afn, bfn[4];
        afn.q[0] = *(const float4*)(aRow + kn);
        afn.q[1] = *(const float4*)(aRow + kn + 16);
        __builtin_prefetch(aRow + kn + 32, 0, 0);        // global_prefetch_b8
#pragma unroll
        for (int j = 0; j < 4; ++j) {
            const _Float16* bp = Bw + (size_t)(colBase + 16 * j) * K + kn + kOffB;
            bfn[j].q[0] = *(const float4*)(bp);
            bfn[j].q[1] = *(const float4*)(bp + 16);
        }
        // ---- consume current fragments
#pragma unroll
        for (int j = 0; j < 4; ++j) {
            acc[j] = __builtin_amdgcn_wmma_f32_16x16x32_f16(
                false, af.v, false, bf[j].v, (short)0, acc[j], false, false);
        }
        // ---- rotate buffers
        af = afn;
#pragma unroll
        for (int j = 0; j < 4; ++j) bf[j] = bfn[j];
    }
    // ---- epilogue: final K slice
#pragma unroll
    for (int j = 0; j < 4; ++j) {
        acc[j] = __builtin_amdgcn_wmma_f32_16x16x32_f16(
            false, af.v, false, bf[j].v, (short)0, acc[j], false, false);
    }

    // C/D layout: VGPR r -> row = 8*halfSel + r, col = lane15
    const int rowBase = tileM + halfSel * 8;
#pragma unroll
    for (int j = 0; j < 4; ++j) {
        const int col = tileN + 16 * j + lane15;
#pragma unroll
        for (int r = 0; r < 8; ++r) {
            const int row = rowBase + r;
            float v = acc[j][r];
            if (MODE == 1) {                               // softplus(acc + b_dt[col])
                v += extra[col];
                v = (v > 20.0f) ? v : log1pf(__expf(v));
            } else if (MODE == 2) {                        // + residual x
                v += extra[(size_t)row * N + col];
            }
            C[(size_t)row * N + col] = v;
        }
    }
}

// ---------------------------------------------------------------------------
// Causal depthwise conv (D_CONV=4) + SiLU. xm = xz[:, 0:D_INNER].
// Writes u in f32 (scan) and f16 (x-proj GEMM input).
// ---------------------------------------------------------------------------
__global__ __launch_bounds__(256) void conv_silu_kernel(const float* __restrict__ xz,
                                                        const float* __restrict__ cw,
                                                        const float* __restrict__ cb,
                                                        float* __restrict__ u_f,
                                                        _Float16* __restrict__ u_h) {
    const int gid = blockIdx.x * 256 + threadIdx.x;      // over ROWS * D_INNER
    const int d   = gid & (D_INNER - 1);
    const int row = gid >> 11;                           // b*L + t
    const int t   = row & (L_SZ - 1);
    float acc = cb[d];
#pragma unroll
    for (int j = 0; j < D_CONV; ++j) {
        const int tt = t - (D_CONV - 1) + j;
        if (tt >= 0)
            acc = fmaf(cw[d * D_CONV + j],
                       xz[(size_t)(row - (D_CONV - 1) + j) * (2 * D_INNER) + d], acc);
    }
    const float s = acc * fast_sigmoid(acc);
    u_f[gid] = s;
    u_h[gid] = (_Float16)s;
}

// ---------------------------------------------------------------------------
// Extract dt (cols 0..63 of x_dbl) -> f16 for the W_dt GEMM
// ---------------------------------------------------------------------------
__global__ __launch_bounds__(256) void extract_dt_kernel(const float* __restrict__ xdbl,
                                                         _Float16* __restrict__ dt) {
    const int i   = blockIdx.x * 256 + threadIdx.x;      // ROWS * DT_RANK
    const int row = i >> 6;
    const int c   = i & 63;
    dt[i] = (_Float16)xdbl[(size_t)row * XPROJ_N + c];
}

// ---------------------------------------------------------------------------
// Selective scan. One thread per (b, d) channel: h[64] + A[64] live in VGPRs,
// B_t / C_t broadcast via LDS, 64-wide unrolled recurrence per step.
// grid = (D_INNER/256, B), block = 256
// ---------------------------------------------------------------------------
__global__ __launch_bounds__(256)
void scan_kernel(const float* __restrict__ delta, const float* __restrict__ u,
                 const float* __restrict__ xdbl, const float* __restrict__ A_log,
                 const float* __restrict__ Dp, float* __restrict__ y) {
    __shared__ float sB[D_STATE];
    __shared__ float sC[D_STATE];
    const int b = blockIdx.y;
    const int d = blockIdx.x * 256 + threadIdx.x;

    float a[D_STATE], h[D_STATE];
#pragma unroll
    for (int n = 0; n < D_STATE; ++n) {
        a[n] = -__expf(A_log[(size_t)d * D_STATE + n]);
        h[n] = 0.f;
    }
    const float Dd = Dp[d];

    for (int t = 0; t < L_SZ; ++t) {
        __syncthreads();                                 // protect sB/sC reuse
        const size_t row = (size_t)(b * L_SZ + t);
        if (threadIdx.x < D_STATE)
            sB[threadIdx.x] = xdbl[row * XPROJ_N + DT_RANK + threadIdx.x];
        else if (threadIdx.x < 2 * D_STATE)
            sC[threadIdx.x - D_STATE] = xdbl[row * XPROJ_N + DT_RANK + D_STATE + (threadIdx.x - D_STATE)];
        __syncthreads();

        const float dt = delta[row * D_INNER + d];
        const float ut = u[row * D_INNER + d];
        const float du = dt * ut;
        float yt = 0.f;
#pragma unroll
        for (int n = 0; n < D_STATE; ++n) {
            const float dA = __expf(dt * a[n]);          // v_exp_f32 chain, ILP=64
            h[n] = fmaf(h[n], dA, du * sB[n]);
            yt   = fmaf(h[n], sC[n], yt);
        }
        y[row * D_INNER + d] = fmaf(ut, Dd, yt);
    }
}

// ---------------------------------------------------------------------------
// Gate: y *= silu(z), z = xz[:, D_INNER:]; write f16 for out-proj GEMM
// ---------------------------------------------------------------------------
__global__ __launch_bounds__(256) void gate_kernel(const float* __restrict__ y,
                                                   const float* __restrict__ xz,
                                                   _Float16* __restrict__ yh) {
    const int i   = blockIdx.x * 256 + threadIdx.x;      // ROWS * D_INNER
    const int row = i >> 11;
    const int d   = i & (D_INNER - 1);
    const float z = xz[(size_t)row * (2 * D_INNER) + D_INNER + d];
    yh[i] = (_Float16)(y[i] * (z * fast_sigmoid(z)));
}

// ---------------------------------------------------------------------------
// Launch
// ---------------------------------------------------------------------------
extern "C" void kernel_launch(void* const* d_in, const int* in_sizes, int n_in,
                              void* d_out, int out_size, void* d_ws, size_t ws_size,
                              hipStream_t stream) {
    const float* x      = (const float*)d_in[0];
    const float* norm_w = (const float*)d_in[1];
    const float* W_in   = (const float*)d_in[2];
    const float* conv_w = (const float*)d_in[3];
    const float* conv_b = (const float*)d_in[4];
    const float* W_x    = (const float*)d_in[5];
    const float* W_dt   = (const float*)d_in[6];
    const float* b_dt   = (const float*)d_in[7];
    const float* A_log  = (const float*)d_in[8];
    const float* Dp     = (const float*)d_in[9];
    const float* W_out  = (const float*)d_in[10];
    float* out = (float*)d_out;

    // --- workspace carve-up (256B aligned) ---
    char* ws = (char*)d_ws;
    size_t off = 0;
    auto carve = [&](size_t bytes) -> void* {
        void* p = ws + off;
        off = (off + bytes + 255) & ~(size_t)255;
        return p;
    };
    _Float16* xn_h   = (_Float16*)carve((size_t)ROWS * D_MODEL * 2);
    _Float16* win_h  = (_Float16*)carve((size_t)2 * D_INNER * D_MODEL * 2);
    _Float16* wx_h   = (_Float16*)carve((size_t)XPROJ_N * D_INNER * 2);
    _Float16* wdt_h  = (_Float16*)carve((size_t)D_INNER * DT_RANK * 2);
    _Float16* wout_h = (_Float16*)carve((size_t)D_MODEL * D_INNER * 2);
    float*    xz     = (float*)carve((size_t)ROWS * 2 * D_INNER * 4);
    float*    u_f    = (float*)carve((size_t)ROWS * D_INNER * 4);
    _Float16* u_h    = (_Float16*)carve((size_t)ROWS * D_INNER * 2);
    float*    xdbl   = (float*)carve((size_t)ROWS * XPROJ_N * 4);
    _Float16* dt_h   = (_Float16*)carve((size_t)ROWS * DT_RANK * 2);
    float*    dlt    = (float*)carve((size_t)ROWS * D_INNER * 4);
    float*    y_f    = (float*)carve((size_t)ROWS * D_INNER * 4);
    _Float16* y_h    = (_Float16*)carve((size_t)ROWS * D_INNER * 2);
    (void)ws_size; (void)n_in; (void)in_sizes; (void)out_size;

    // --- weight converts to f16 ---
    {
        int n;
        n = 2 * D_INNER * D_MODEL;
        cvt_f16_kernel<<<(n + 255) / 256, 256, 0, stream>>>(W_in, win_h, n);
        n = XPROJ_N * D_INNER;
        cvt_f16_kernel<<<(n + 255) / 256, 256, 0, stream>>>(W_x, wx_h, n);
        n = D_INNER * DT_RANK;
        cvt_f16_kernel<<<(n + 255) / 256, 256, 0, stream>>>(W_dt, wdt_h, n);
        n = D_MODEL * D_INNER;
        cvt_f16_kernel<<<(n + 255) / 256, 256, 0, stream>>>(W_out, wout_h, n);
    }

    // --- RMSNorm -> xn_h ---
    rmsnorm_kernel<<<ROWS, 256, 0, stream>>>(x, norm_w, xn_h);

    // --- in-projection: xz = xn @ W_in^T  [8192 x 4096, K=1024] ---
    gemm_wmma_f16<0><<<dim3((2 * D_INNER) / 64, ROWS / 64), 128, 0, stream>>>(
        xn_h, win_h, xz, nullptr, ROWS, 2 * D_INNER, D_MODEL);

    // --- causal depthwise conv + SiLU -> u ---
    conv_silu_kernel<<<(ROWS * D_INNER) / 256, 256, 0, stream>>>(xz, conv_w, conv_b, u_f, u_h);

    // --- x-projection: x_dbl = u @ W_x^T  [8192 x 192, K=2048] ---
    gemm_wmma_f16<0><<<dim3(XPROJ_N / 64, ROWS / 64), 128, 0, stream>>>(
        u_h, wx_h, xdbl, nullptr, ROWS, XPROJ_N, D_INNER);

    // --- dt slice -> f16 ---
    extract_dt_kernel<<<(ROWS * DT_RANK) / 256, 256, 0, stream>>>(xdbl, dt_h);

    // --- delta = softplus(dt @ W_dt^T + b_dt)  [8192 x 2048, K=64] ---
    gemm_wmma_f16<1><<<dim3(D_INNER / 64, ROWS / 64), 128, 0, stream>>>(
        dt_h, wdt_h, dlt, b_dt, ROWS, D_INNER, DT_RANK);

    // --- selective scan -> y_f ---
    scan_kernel<<<dim3(D_INNER / 256, B_SZ), 256, 0, stream>>>(dlt, u_f, xdbl, A_log, Dp, y_f);

    // --- gate: y *= silu(z) -> y_h ---
    gate_kernel<<<(ROWS * D_INNER) / 256, 256, 0, stream>>>(y_f, xz, y_h);

    // --- out-projection + residual: out = y @ W_out^T + x  [8192 x 1024, K=2048] ---
    gemm_wmma_f16<2><<<dim3(D_MODEL / 64, ROWS / 64), 128, 0, stream>>>(
        y_h, wout_h, out, x, ROWS, D_MODEL, D_INNER);
}
